// AttnRefine_39290360823953
// MI455X (gfx1250) — matrix operand order repaired and checked by reading, not compiled, not run on hardware
//
#include <hip/hip_runtime.h>
#include <hip/hip_bf16.h>

// ============================================================================
// MI455X (gfx1250) implementation of the AttnRefine network.
//  - All intermediate activations: bf16, NHWC, channels padded to mult-of-32.
//  - 3x3 convs: implicit GEMM on v_wmma_f32_16x16x32_bf16 (wave32),
//      templated on <CIN, STRIDE> so the K loop (9 taps x CIN/32 chunks) is
//      fully unrolled: LDS A-fragment loads become ds_load_b128 with
//      immediate offsets, B loads are contiguous 32B per lane, zero division.
//  - LDS halo fill: 16B groups via global_load_async_to_lds_b128 when the
//    toolchain exposes it (s_wait_asynccnt 0 before the barrier), else
//    vectorized v8bf load/store.
// ============================================================================

typedef __bf16 bf16;
typedef __attribute__((ext_vector_type(16))) __bf16 v16bf;
typedef __attribute__((ext_vector_type(8)))  __bf16 v8bf;
typedef __attribute__((ext_vector_type(8)))  float  v8f;
typedef int v4i __attribute__((vector_size(16)));

#if defined(__HIP_DEVICE_COMPILE__) && \
    __has_builtin(__builtin_amdgcn_global_load_async_to_lds_b128)
#define ASYNC_LDS 1
typedef __attribute__((address_space(1))) v4i* g_v4i_ptr;
typedef __attribute__((address_space(3))) v4i* l_v4i_ptr;
#else
#define ASYNC_LDS 0
#endif

// ---------------------------------------------------------------------------
// Weight packing: OIHW fp32 -> bf16 B-fragment layout.
// K order: k = tap*CinPad + ci  (tap = ky*3+kx).
// B-matrix (32x16 bf16) lane layout: element i of lane l -> K = i + 16*(l>=16),
// column N = l & 15.  Packed as [tile][chunk][lane][16].
// ---------------------------------------------------------------------------
__global__ void pack_weights_kernel(const float* __restrict__ w,
                                    bf16* __restrict__ wp,
                                    int CoutReal, int CinReal, int CinPad,
                                    int tiles) {
  const int nchunks = (9 * CinPad) >> 5;
  const size_t total = (size_t)tiles * nchunks * 512;
  size_t idx = (size_t)blockIdx.x * blockDim.x + threadIdx.x;
  if (idx >= total) return;
  int i     = (int)(idx & 15);
  int lane  = (int)((idx >> 4) & 31);
  int chunk = (int)((idx >> 9) % (size_t)nchunks);
  int tile  = (int)(idx / ((size_t)nchunks << 9));
  int koff  = i + ((lane >> 4) << 4);
  int k     = (chunk << 5) + koff;
  int t     = k / CinPad;
  int ci    = k - t * CinPad;
  int oc    = tile * 16 + (lane & 15);
  float v = 0.0f;
  if (oc < CoutReal && ci < CinReal)
    v = w[((size_t)oc * CinReal + ci) * 9 + t];
  wp[idx] = (bf16)v;
}

// ---------------------------------------------------------------------------
// fp32 NCHW (3 channels out of srcC, starting at c0) -> bf16 NHWC, C padded
// to 32 with zeros.  Fixed 16 x 128 x 256 images.
// ---------------------------------------------------------------------------
__global__ void pad_to_nhwc_kernel(const float* __restrict__ src, int srcC,
                                   int c0, bf16* __restrict__ dst) {
  const size_t total = (size_t)16 * 128 * 256 * 32;
  size_t idx = (size_t)blockIdx.x * blockDim.x + threadIdx.x;
  if (idx >= total) return;
  int c = (int)(idx & 31);
  size_t p = idx >> 5;
  int x = (int)(p % 256); p /= 256;
  int y = (int)(p % 128);
  int b = (int)(p / 128);
  float v = 0.0f;
  if (c < 3)
    v = src[(((size_t)b * srcC + (c0 + c)) * 128 + y) * 256 + x];
  dst[idx] = (bf16)v;
}

// ---------------------------------------------------------------------------
// Implicit-GEMM 3x3 conv (pad=1) on WMMA bf16, templated on <CIN, STRIDE>.
//  grid  = (Wout/16, Hout, B);   blockDim.x = 2*CoutPad  (one wave / oc-tile)
//  dynamic LDS: 3 * (15*STRIDE+3) * CIN bf16 halo tile.
//  Epilogue fuses bias, optional residual add, optional LeakyReLU(0.2).
//  If outF32 != null: write fp32 NCHW (oc < CoutReal only) — final layer.
// ---------------------------------------------------------------------------
template <int CIN, int STRIDE>
__global__ void conv3x3_wmma_kernel(
    const bf16* __restrict__ in, int Hin, int Win,
    const bf16* __restrict__ wpack, const float* __restrict__ bias,
    bf16* __restrict__ out, int Hout, int Wout, int Cout,
    int doRelu, int CoutReal,
    const bf16* __restrict__ resid, float* __restrict__ outF32) {
  constexpr int Wt  = 15 * STRIDE + 3;   // staged input width
  constexpr int CB  = CIN / 32;          // 32-wide K chunks per tap
  constexpr int C8  = CIN / 8;           // 16-byte groups per pixel
  constexpr int GROUPS = 3 * Wt * C8;    // 16-byte groups in the halo tile

  extern __shared__ char smem_raw[];
  bf16* sm = reinterpret_cast<bf16*>(smem_raw);

  const int b   = blockIdx.z;
  const int oy  = blockIdx.y;
  const int ox0 = blockIdx.x << 4;
  const int tid = threadIdx.x;

  // ---- cooperative halo fill, 16 bytes per iteration per thread
  for (int g = tid; g < GROUPS; g += blockDim.x) {
    int ci8 = g % C8;
    int r   = g / C8;
    int xx  = r % Wt;
    int ry  = r / Wt;
    int iy  = oy * STRIDE + ry - 1;
    int ix  = ox0 * STRIDE + xx - 1;
    bool ib = (iy >= 0) && (iy < Hin) && (ix >= 0) && (ix < Win);
    const bf16* src =
        in + ((((size_t)b * Hin + iy) * Win + ix) * CIN + (size_t)ci8 * 8);
#if ASYNC_LDS
    if (ib) {
      __builtin_amdgcn_global_load_async_to_lds_b128(
          (g_v4i_ptr)src, (l_v4i_ptr)&sm[(size_t)g * 8], 0, 0);
    } else {
      v8bf z = {};
      *reinterpret_cast<v8bf*>(&sm[(size_t)g * 8]) = z;
    }
#else
    v8bf v = {};
    if (ib) v = *reinterpret_cast<const v8bf*>(src);
    *reinterpret_cast<v8bf*>(&sm[(size_t)g * 8]) = v;
#endif
  }
#if ASYNC_LDS
#if __has_builtin(__builtin_amdgcn_s_wait_asynccnt)
  __builtin_amdgcn_s_wait_asynccnt(0);
#else
  asm volatile("s_wait_asynccnt 0" ::: "memory");
#endif
#endif
  __syncthreads();

  const int wave = tid >> 5;
  const int lane = tid & 31;
  const int l16  = (lane >> 4) & 1;   // lane-half selects K interleave
  const int m    = lane & 15;         // A row = output pixel within strip

  v8f acc = {};
  // per-lane base of packed weights: [wave tile][chunk][lane][16]
  const bf16* wb = wpack + (size_t)wave * (9 * CB) * 512 + (size_t)lane * 16;
  __builtin_prefetch(wb, 0, 1);

  const int abase = m * STRIDE * CIN + (l16 << 4);

#pragma unroll
  for (int t = 0; t < 9; ++t) {
    const int dy = t / 3, dx = t % 3;               // compile-time constants
    const int pixoff = (dy * Wt + dx) * CIN + abase;
#pragma unroll
    for (int cb = 0; cb < CB; ++cb) {
      // A fragment (16x32 bf16): lanes 0-15 hold K{0-15},
      //                          lanes 16-31 hold K{16-31}
      const int off0 = pixoff + cb * 32;
      v8bf a_lo = *reinterpret_cast<const v8bf*>(&sm[off0]);
      v8bf a_hi = *reinterpret_cast<const v8bf*>(&sm[off0 + 8]);
      v16bf a;
#pragma unroll
      for (int i = 0; i < 8; ++i) { a[i] = a_lo[i]; a[i + 8] = a_hi[i]; }

      // B fragment: 32 contiguous bytes per lane
      const v16bf bfrag =
          *reinterpret_cast<const v16bf*>(wb + (size_t)(t * CB + cb) * 512);

      acc = __builtin_amdgcn_wmma_f32_16x16x32_bf16(
          false, a, false, bfrag, (short)0, acc, false, false);
    }
  }

  // ---- epilogue: D layout -> element j of lane = (M = j + 8*l16, N = lane&15)
  const int n  = lane & 15;
  const int oc = wave * 16 + n;
  const float bv = (oc < CoutReal) ? bias[oc] : 0.0f;
#pragma unroll
  for (int j = 0; j < 8; ++j) {
    int mm = j + (l16 << 3);
    int ox = ox0 + mm;
    float v = acc[j] + bv;
    size_t oidx = (((size_t)b * Hout + oy) * Wout + ox) * (size_t)Cout + oc;
    if (resid) v += (float)resid[oidx];
    if (doRelu) v = (v >= 0.0f) ? v : 0.2f * v;
    if (outF32) {
      if (oc < CoutReal)
        outF32[(((size_t)b * CoutReal + oc) * Hout + oy) * Wout + ox] = v;
    } else {
      out[oidx] = (bf16)v;
    }
  }
}

// ---------------------------------------------------------------------------
// Correlation attention over 5x5 window (pad value 1e-6), softmax, argmax
// offset, softmax-weighted aggregation.  One thread per (b, n, h, w).
// Writes cat192 = concat(cpat[64], agg[128]) NHWC and offsets (fp32) to d_out.
// ---------------------------------------------------------------------------
__global__ void corr_attn_kernel(const bf16* __restrict__ cpat,
                                 const bf16* __restrict__ nb1,
                                 const bf16* __restrict__ nb2,
                                 bf16* __restrict__ cat,
                                 float* __restrict__ offs) {
  const int H = 32, W = 64, C = 64;
  int idx = blockIdx.x * blockDim.x + threadIdx.x;
  if (idx >= 16 * 2 * H * W) return;
  int w = idx & 63;
  int h = (idx >> 6) & 31;
  int n = (idx >> 11) & 1;
  int b = idx >> 12;

  const bf16* nb = n ? nb2 : nb1;
  const bf16* cp = cpat + (((size_t)b * H + h) * W + w) * C;

  float corr[25];
#pragma unroll
  for (int k = 0; k < 25; ++k) {
    int dy = k / 5 - 2, dx = k % 5 - 2;
    int yy = h + dy, xx = w + dx;
    bool ib = (yy >= 0 && yy < H && xx >= 0 && xx < W);
    const bf16* wv = ib ? (nb + (((size_t)b * H + yy) * W + xx) * C) : nullptr;
    float dot = 0.0f, ss = 0.0f;
    for (int c = 0; c < C; ++c) {
      float v = ib ? (float)wv[c] : 1e-6f;
      dot += v * (float)cp[c];
      ss  += v * v;
    }
    corr[k] = dot * rsqrtf(ss);
  }

  float mx = corr[0]; int am = 0;
#pragma unroll
  for (int k = 1; k < 25; ++k)
    if (corr[k] > mx) { mx = corr[k]; am = k; }

  float sm[25], den = 0.0f;
#pragma unroll
  for (int k = 0; k < 25; ++k) { sm[k] = __expf(corr[k] - mx); den += sm[k]; }
  const float rden = 1.0f / den;

  float agg[64];
  for (int c = 0; c < C; ++c) agg[c] = 0.0f;
  for (int k = 0; k < 25; ++k) {
    int dy = k / 5 - 2, dx = k % 5 - 2;
    int yy = h + dy, xx = w + dx;
    float wk = sm[k] * rden;
    bool ib = (yy >= 0 && yy < H && xx >= 0 && xx < W);
    const bf16* wv = ib ? (nb + (((size_t)b * H + yy) * W + xx) * C) : nullptr;
    for (int c = 0; c < C; ++c) {
      float v = ib ? (float)wv[c] : 1e-6f;
      agg[c] += wk * v;
    }
  }

  bf16* co = cat + (((size_t)b * H + h) * W + w) * 192;
  if (n == 0)
    for (int c = 0; c < C; ++c) co[c] = cp[c];
  for (int c = 0; c < C; ++c) co[64 + n * 64 + c] = (bf16)agg[c];

  offs[((((size_t)b * 2 + n) * 2 + 0) * H + h) * W + w] = (float)(am / 5 - 2);
  offs[((((size_t)b * 2 + n) * 2 + 1) * H + h) * W + w] = (float)(am % 5 - 2);
}

// ---------------------------------------------------------------------------
// Bilinear x2 upsample, align_corners=true, NHWC bf16.  Thread per (b,y,x).
// ---------------------------------------------------------------------------
__global__ void up2_bilinear_kernel(const bf16* __restrict__ in, int Hin,
                                    int Win, int C, bf16* __restrict__ out) {
  const int Hout = 2 * Hin, Wout = 2 * Win;
  int idx = blockIdx.x * blockDim.x + threadIdx.x;
  if (idx >= 16 * Hout * Wout) return;
  int x = idx % Wout;
  int t = idx / Wout;
  int y = t % Hout;
  int b = t / Hout;
  float py = (float)y * (float)(Hin - 1) / (float)(Hout - 1);
  float px = (float)x * (float)(Win - 1) / (float)(Wout - 1);
  int y0 = (int)py, x0 = (int)px;
  int y1 = min(y0 + 1, Hin - 1), x1 = min(x0 + 1, Win - 1);
  float fy = py - (float)y0, fx = px - (float)x0;
  const bf16* p00 = in + (((size_t)b * Hin + y0) * Win + x0) * C;
  const bf16* p01 = in + (((size_t)b * Hin + y0) * Win + x1) * C;
  const bf16* p10 = in + (((size_t)b * Hin + y1) * Win + x0) * C;
  const bf16* p11 = in + (((size_t)b * Hin + y1) * Win + x1) * C;
  bf16* o = out + (((size_t)b * Hout + y) * Wout + x) * C;
  for (int c = 0; c < C; ++c) {
    float v = (1.0f - fy) * ((1.0f - fx) * (float)p00[c] + fx * (float)p01[c]) +
              fy * ((1.0f - fx) * (float)p10[c] + fx * (float)p11[c]);
    o[c] = (bf16)v;
  }
}

__global__ void add_bf16_kernel(const bf16* __restrict__ a,
                                const bf16* __restrict__ b,
                                bf16* __restrict__ o, size_t n) {
  size_t i = (size_t)blockIdx.x * blockDim.x + threadIdx.x;
  if (i < n) o[i] = (bf16)((float)a[i] + (float)b[i]);
}

// ===========================================================================
extern "C" void kernel_launch(void* const* d_in, const int* in_sizes, int n_in,
                              void* d_out, int out_size, void* d_ws,
                              size_t ws_size, hipStream_t stream) {
  (void)in_sizes; (void)n_in; (void)out_size; (void)ws_size;
  const float* coarse = (const float*)d_in[0];
  const float* neigh  = (const float*)d_in[1];
  auto P = [&](int i) { return (const float*)d_in[i]; };

  // ---- bump allocator over workspace
  char* ws = (char*)d_ws;
  size_t off = 0;
  auto alloc = [&](size_t bytes) -> void* {
    void* p = ws + off;
    off += (bytes + 255) & ~(size_t)255;
    return p;
  };

  const size_t EF32 = (size_t)16 * 128 * 256 * 32;  // full-res, 32ch (elems)
  const size_t EH64 = (size_t)16 * 64 * 128 * 64;   // half-res, 64ch
  const size_t EQ64 = (size_t)16 * 32 * 64 * 64;    // quarter-res, 64ch

  bf16* A    = (bf16*)alloc(EF32 * 2);
  bf16* e1c  = (bf16*)alloc(EF32 * 2);
  bf16* e1s  = (bf16*)alloc(EF32 * 2);
  bf16* th   = (bf16*)alloc(EH64 * 2);
  bf16* e2c  = (bf16*)alloc(EH64 * 2);
  bf16* e2s  = (bf16*)alloc(EH64 * 2);
  bf16* tq   = (bf16*)alloc(EQ64 * 2);
  bf16* cpat = (bf16*)alloc(EQ64 * 2);
  bf16* nb1  = (bf16*)alloc(EQ64 * 2);
  bf16* nb2  = (bf16*)alloc(EQ64 * 2);
  bf16* cat  = (bf16*)alloc((size_t)16 * 32 * 64 * 192 * 2);
  bf16* x128 = (bf16*)alloc((size_t)16 * 32 * 64 * 128 * 2);
  bf16* x64a = (bf16*)alloc(EQ64 * 2);
  bf16* x64b = (bf16*)alloc(EQ64 * 2);
  bf16* d2a  = (bf16*)alloc(EH64 * 2);
  bf16* d2b  = (bf16*)alloc(EH64 * 2);
  bf16* d1a  = (bf16*)alloc((size_t)16 * 128 * 256 * 64 * 2);
  bf16* d1b  = (bf16*)alloc(EF32 * 2);
  bf16* d1c  = (bf16*)alloc(EF32 * 2);

  // ---- layer table: {w idx, b idx, CoutReal, CinReal, CinPad, CoutPad}
  struct Layer { int wi, bi, CoR, CiR, CiP, CoP; };
  const Layer Ls[18] = {
      {2, 3, 32, 3, 32, 32},       // 0  enc1
      {4, 5, 64, 32, 32, 64},      // 1  enc2a (s2)
      {6, 7, 64, 64, 64, 64},      // 2  enc2b
      {8, 9, 64, 64, 64, 64},      // 3  enc3a (s2)
      {10, 11, 64, 64, 64, 64},    // 4  enc3b (no relu)
      {12, 13, 128, 192, 192, 128},// 5  dec3a
      {14, 15, 64, 128, 128, 64},  // 6  dec3b
      {16, 17, 64, 64, 64, 64},    // 7  rb1 conv1
      {18, 19, 64, 64, 64, 64},    // 8  rb1 conv2 (+res)
      {20, 21, 64, 64, 64, 64},    // 9  rb2 conv1
      {22, 23, 64, 64, 64, 64},    // 10 rb2 conv2 (+res)
      {24, 25, 64, 64, 64, 64},    // 11 dec2a
      {26, 27, 64, 64, 64, 64},    // 12 dec2b
      {28, 29, 32, 64, 64, 32},    // 13 dec1a
      {30, 31, 32, 32, 32, 32},    // 14 dec1b
      {32, 33, 32, 32, 32, 32},    // 15 out_rb conv1
      {34, 35, 32, 32, 32, 32},    // 16 out_rb conv2 (+res)
      {36, 37, 3, 32, 32, 16},     // 17 out (fp32 NCHW)
  };

  // ---- pack all weights into WMMA B-fragment layout
  bf16* wp[18];
  for (int l = 0; l < 18; ++l) {
    int tiles = Ls[l].CoP / 16;
    int nch   = 9 * Ls[l].CiP / 32;
    size_t elems = (size_t)tiles * nch * 512;
    wp[l] = (bf16*)alloc(elems * 2);
    unsigned blocks = (unsigned)((elems + 255) / 256);
    pack_weights_kernel<<<blocks, 256, 0, stream>>>(
        P(Ls[l].wi), wp[l], Ls[l].CoR, Ls[l].CiR, Ls[l].CiP, tiles);
  }

  auto conv = [&](int l, const bf16* in, int Hin, int Win, bf16* out, int Hout,
                  int Wout, int stride, int relu, const bf16* resid,
                  float* f32out) {
    const Layer& L = Ls[l];
    dim3 grid(Wout / 16, Hout, 16);
    int threads = 2 * L.CoP;  // 32 threads (one wave) per 16-oc tile
    size_t shmem = (size_t)3 * (15 * stride + 3) * L.CiP * 2;
#define LAUNCH_CONV(CI, ST)                                                    \
  conv3x3_wmma_kernel<CI, ST><<<grid, threads, shmem, stream>>>(               \
      in, Hin, Win, wp[l], P(L.bi), out, Hout, Wout, L.CoP, relu, L.CoR,       \
      resid, f32out)
    if (stride == 1) {
      switch (L.CiP) {
        case 32:  LAUNCH_CONV(32, 1);  break;
        case 64:  LAUNCH_CONV(64, 1);  break;
        case 128: LAUNCH_CONV(128, 1); break;
        default:  LAUNCH_CONV(192, 1); break;
      }
    } else {
      switch (L.CiP) {
        case 32:  LAUNCH_CONV(32, 2);  break;
        default:  LAUNCH_CONV(64, 2);  break;
      }
    }
#undef LAUNCH_CONV
  };

  // ---- encoder (shared weights), run for coarse + both neighbors
  auto encode = [&](const float* src, int srcC, int c0, bf16* e1o, bf16* e2o,
                    bf16* e3o) {
    {
      unsigned blocks = (unsigned)((EF32 + 255) / 256);
      pad_to_nhwc_kernel<<<blocks, 256, 0, stream>>>(src, srcC, c0, A);
    }
    conv(0, A,   128, 256, e1o, 128, 256, 1, 1, nullptr, nullptr);
    conv(1, e1o, 128, 256, th,  64, 128,  2, 1, nullptr, nullptr);
    conv(2, th,  64, 128,  e2o, 64, 128,  1, 1, nullptr, nullptr);
    conv(3, e2o, 64, 128,  tq,  32, 64,   2, 1, nullptr, nullptr);
    conv(4, tq,  32, 64,   e3o, 32, 64,   1, 0, nullptr, nullptr);
  };
  encode(coarse, 3, 0, e1c, e2c, cpat);
  encode(neigh, 6, 0, e1s, e2s, nb1);
  encode(neigh, 6, 3, e1s, e2s, nb2);

  // ---- correlation attention -> cat192 + offsets (second output)
  float* out_img = (float*)d_out;
  float* out_off = out_img + (size_t)16 * 3 * 128 * 256;
  {
    int total = 16 * 2 * 32 * 64;
    corr_attn_kernel<<<(total + 255) / 256, 256, 0, stream>>>(
        cpat, nb1, nb2, cat, out_off);
  }

  // ---- decoder
  conv(5, cat,  32, 64, x128, 32, 64, 1, 1, nullptr, nullptr);
  conv(6, x128, 32, 64, x64a, 32, 64, 1, 1, nullptr, nullptr);
  conv(7, x64a, 32, 64, x64b, 32, 64, 1, 1, nullptr, nullptr);   // rb1
  conv(8, x64b, 32, 64, x64a, 32, 64, 1, 0, x64a, nullptr);
  conv(9, x64a, 32, 64, x64b, 32, 64, 1, 1, nullptr, nullptr);   // rb2
  conv(10, x64b, 32, 64, x64a, 32, 64, 1, 0, x64a, nullptr);

  {
    int total = 16 * 64 * 128;
    up2_bilinear_kernel<<<(total + 255) / 256, 256, 0, stream>>>(
        x64a, 32, 64, 64, d2a);
  }
  conv(11, d2a, 64, 128, d2b, 64, 128, 1, 1, nullptr, nullptr);
  conv(12, d2b, 64, 128, d2a, 64, 128, 1, 1, nullptr, nullptr);
  {
    add_bf16_kernel<<<(unsigned)((EH64 + 255) / 256), 256, 0, stream>>>(
        d2a, e2c, d2b, EH64);
  }
  {
    int total = 16 * 128 * 256;
    up2_bilinear_kernel<<<(total + 255) / 256, 256, 0, stream>>>(
        d2b, 64, 128, 64, d1a);
  }
  conv(13, d1a, 128, 256, d1b, 128, 256, 1, 1, nullptr, nullptr);
  conv(14, d1b, 128, 256, d1c, 128, 256, 1, 1, nullptr, nullptr);
  {
    add_bf16_kernel<<<(unsigned)((EF32 + 255) / 256), 256, 0, stream>>>(
        d1c, e1c, d1b, EF32);
  }
  conv(15, d1b, 128, 256, d1c, 128, 256, 1, 1, nullptr, nullptr); // out_rb
  conv(16, d1c, 128, 256, d1b, 128, 256, 1, 0, d1b, nullptr);
  conv(17, d1b, 128, 256, d1c, 128, 256, 1, 0, nullptr, out_img); // -> fp32
}
